// TwoSimplicialAttention_47639777247276
// MI455X (gfx1250) — matrix operand order repaired and checked by reading, not compiled, not run on hardware
//
#include <hip/hip_runtime.h>
#include <stdint.h>

typedef __attribute__((ext_vector_type(16))) _Float16 v16h;
typedef __attribute__((ext_vector_type(8)))  float    v8f;

union FragAB { v16h h; uint32_t u[8]; };
union FragC  { v8f  v; float    f[8]; };

static constexpr int Hh  = 4;
static constexpr int T   = 256;
static constexpr int Dd  = 64;
static constexpr int IPB = 4;   // i's per block

__device__ __forceinline__ uint32_t pkf16(float lo, float hi) {
    union { _Float16 h[2]; uint32_t u; } t;
    t.h[0] = (_Float16)lo; t.h[1] = (_Float16)hi;
    return t.u;
}

__device__ __forceinline__ float wave_max(float x) {
    #pragma unroll
    for (int o = 16; o > 0; o >>= 1) x = fmaxf(x, __shfl_xor(x, o, 32));
    return x;
}
__device__ __forceinline__ float wave_sum(float x) {
    #pragma unroll
    for (int o = 16; o > 0; o >>= 1) x += __shfl_xor(x, o, 32);
    return x;
}

__global__ __launch_bounds__(256) void tsattn_kernel(
    const float* __restrict__ qg,  const float* __restrict__ k1g,
    const float* __restrict__ v1g, const float* __restrict__ k2g,
    const float* __restrict__ v2g, const float* __restrict__ bug,
    const float* __restrict__ bvg, const float* __restrict__ bwg,
    const float* __restrict__ bgg, const uint8_t* __restrict__ maskg,
    float* __restrict__ outg)
{
    __shared__ _Float16 k1_s [T * Dd];   // k1[j][d]  f16       32KB
    __shared__ _Float16 v1t_s[Dd * T];   // v1^T[d][j] f16      32KB
    __shared__ _Float16 qk2_s[T * Dd];   // (q_i*k2)[k][d] f16  32KB
    __shared__ float    maskadd_s[T];
    __shared__ float    out_acc[Dd];
    __shared__ float    wm_s[8];         // per-wave running max
    __shared__ float    ws_s[8];         // per-wave partial sum

    const int tid  = threadIdx.x;
    const int wave = tid >> 5;
    const int lane = tid & 31;
    const int lhi  = (lane >= 16) ? 1 : 0;
    const int l15  = lane & 15;

    const int h     = blockIdx.x / (T / IPB);
    const int i0    = (blockIdx.x % (T / IPB)) * IPB;
    const int hbase = h * T * Dd;

    // ---- stage per-head constants in LDS (f16) ----
    for (int idx = tid; idx < T * Dd; idx += 256) {
        const int j = idx >> 6, d = idx & 63;
        k1_s[idx]        = (_Float16)k1g[hbase + idx];
        v1t_s[d * T + j] = (_Float16)v1g[hbase + idx];
    }
    if (tid < T) maskadd_s[tid] = maskg[tid] ? -1e30f : 0.0f;
    __syncthreads();

    for (int i = i0; i < i0 + IPB; ++i) {
        const int    ib    = h * T + i;
        const float  gate  = bgg[ib];
        const float* burow = bug + (size_t)ib * T;
        const float* bvrow = bvg + (size_t)ib * T;
        const float* bwh   = bwg + (size_t)h * T * T;

        // qk2[k][d] = q[i][d] * k2[k][d]
        for (int idx = tid; idx < T * Dd; idx += 256) {
            const int d = idx & 63;
            qk2_s[idx] = (_Float16)(qg[(size_t)ib * Dd + d] * k2g[hbase + idx]);
        }
        if (tid < Dd) out_acc[tid] = 0.0f;
        __syncthreads();

        // ---- logits tile: S[j0..j0+15][k0..k0+15] + bias + mask ----
        auto tile_logits = [&](int j0, int k0, float lf[8]) {
            FragC c;
            #pragma unroll
            for (int r = 0; r < 8; ++r) c.f[r] = 0.0f;
            #pragma unroll
            for (int db = 0; db < Dd; db += 32) {
                FragAB a, b;
                #pragma unroll
                for (int r = 0; r < 8; ++r) {   // A: k1[j][d], 16x32 f16 layout
                    const int doff = db + ((r < 4) ? 0 : 16) + (lhi ? 8 : 0) + 2 * (r & 3);
                    a.u[r] = *(const uint32_t*)&k1_s[(j0 + l15) * Dd + doff];
                }
                #pragma unroll
                for (int r = 0; r < 8; ++r) {   // B: qk2^T, 32x16 f16 layout
                    const int doff = db + ((r < 4) ? 0 : 16) + (lhi ? 8 : 0) + 2 * (r & 3);
                    b.u[r] = *(const uint32_t*)&qk2_s[(k0 + l15) * Dd + doff];
                }
                c.v = __builtin_amdgcn_wmma_f32_16x16x32_f16(
                        false, a.h, false, b.h, (short)0, c.v, false, false);
            }
            const int   k    = k0 + l15;
            const float addk = gate * bvrow[k] + maskadd_s[k];
            #pragma unroll
            for (int r = 0; r < 8; ++r) {       // D tile: M=j row r(+8), N=k lane
                const int j = j0 + r + (lhi ? 8 : 0);
                lf[r] = c.f[r] + gate * (burow[j] + bwh[(size_t)j * T + k])
                        + maskadd_s[j] + addk;
            }
        };

        // ===== single pass: online softmax + C2[k,d] += P^T @ v1 =====
        float mcur = -3.0e38f;   // wave-uniform running max
        float lsum = 0.0f;       // per-lane partial exp-sum
        FragC acc2[2][4];
        #pragma unroll
        for (int cc = 0; cc < 2; ++cc)
            #pragma unroll
            for (int dt = 0; dt < 4; ++dt)
                #pragma unroll
                for (int r = 0; r < 8; ++r) acc2[cc][dt].f[r] = 0.0f;

        for (int jt = 0; jt < 16; jt += 2) {
            for (int cc = 0; cc < 2; ++cc) {
                const int k0 = (2 * wave + cc) * 16;
                float l0[8], l1[8];
                tile_logits(jt * 16,      k0, l0);
                tile_logits(jt * 16 + 16, k0, l1);

                // wave-uniform running max update + accumulator rescale
                float tmax = -3.0e38f;
                #pragma unroll
                for (int r = 0; r < 8; ++r) tmax = fmaxf(tmax, fmaxf(l0[r], l1[r]));
                tmax = wave_max(tmax);
                const float mnew = fmaxf(mcur, tmax);
                if (mnew > mcur) {                 // uniform branch
                    const float scale = __expf(mcur - mnew);
                    lsum *= scale;
                    #pragma unroll
                    for (int c2 = 0; c2 < 2; ++c2)
                        #pragma unroll
                        for (int dt = 0; dt < 4; ++dt)
                            #pragma unroll
                            for (int r = 0; r < 8; ++r) acc2[c2][dt].f[r] *= scale;
                    mcur = mnew;
                }

                float p0[8], p1[8];
                #pragma unroll
                for (int r = 0; r < 8; ++r) {
                    p0[r] = __expf(l0[r] - mcur);
                    p1[r] = __expf(l1[r] - mcur);
                    lsum += p0[r] + p1[r];
                }
                // In-register transpose: D-tile (M=j,N=k) pair == A-frag of
                // P^T (M=k, K=32 j's) lane-for-lane; only a VGPR repack.
                FragAB pa;
                #pragma unroll
                for (int r = 0; r < 4; ++r) {
                    pa.u[r]     = pkf16(p0[2 * r], p0[2 * r + 1]);
                    pa.u[r + 4] = pkf16(p1[2 * r], p1[2 * r + 1]);
                }
                #pragma unroll
                for (int dt = 0; dt < 4; ++dt) {
                    FragAB bv;                    // B: v1 (K=j, N=d) from v1^T LDS
                    #pragma unroll
                    for (int r = 0; r < 8; ++r) {
                        const int jh = jt * 16 + ((r < 4) ? 0 : 16) + (lhi ? 8 : 0) + 2 * (r & 3);
                        bv.u[r] = *(const uint32_t*)&v1t_s[(dt * 16 + l15) * T + jh];
                    }
                    acc2[cc][dt].v = __builtin_amdgcn_wmma_f32_16x16x32_f16(
                        false, pa.h, false, bv.h, (short)0, acc2[cc][dt].v, false, false);
                }
            }
        }

        // ===== merge per-wave (m_w, S_w) across the 8 waves =====
        const float wsum = wave_sum(lsum);
        if (lane == 0) { wm_s[wave] = mcur; ws_s[wave] = wsum; }
        __syncthreads();
        float mglob = -3.0e38f;
        #pragma unroll
        for (int w = 0; w < 8; ++w) mglob = fmaxf(mglob, wm_s[w]);
        float Sglob = 0.0f;
        #pragma unroll
        for (int w = 0; w < 8; ++w) Sglob += ws_s[w] * __expf(wm_s[w] - mglob);
        const float factor = __expf(mcur - mglob);       // wave's rescale
        const float inv    = (Sglob > 0.0f) ? (1.0f / Sglob) : 0.0f;

        // out[i][d] = (1/S) * sum_k C2[k][d] * v2[k][d]
        #pragma unroll
        for (int cc = 0; cc < 2; ++cc) {
            #pragma unroll
            for (int dt = 0; dt < 4; ++dt) {
                const int d = dt * 16 + l15;
                float part = 0.0f;
                #pragma unroll
                for (int r = 0; r < 8; ++r) {   // D tile: M=k row r(+8), N=d lane
                    const int k = (2 * wave + cc) * 16 + r + (lhi ? 8 : 0);
                    part += acc2[cc][dt].f[r] * v2g[hbase + k * Dd + d];
                }
                atomicAdd(&out_acc[d], part * factor);   // ds_add_f32
            }
        }
        __syncthreads();
        if (tid < Dd)
            outg[(size_t)ib * Dd + tid] = out_acc[tid] * inv * (maskg[i] ? 0.0f : 1.0f);
        __syncthreads();
    }
}

extern "C" void kernel_launch(void* const* d_in, const int* in_sizes, int n_in,
                              void* d_out, int out_size, void* d_ws, size_t ws_size,
                              hipStream_t stream) {
    (void)in_sizes; (void)n_in; (void)out_size; (void)d_ws; (void)ws_size;
    const float*   q    = (const float*)d_in[0];
    const float*   k1   = (const float*)d_in[1];
    const float*   v1   = (const float*)d_in[2];
    const float*   k2   = (const float*)d_in[3];
    const float*   v2   = (const float*)d_in[4];
    const float*   bu   = (const float*)d_in[5];
    const float*   bv   = (const float*)d_in[6];
    const float*   bw   = (const float*)d_in[7];
    const float*   bg   = (const float*)d_in[8];
    const uint8_t* mask = (const uint8_t*)d_in[9];
    float*         out  = (float*)d_out;

    dim3 grid(Hh * (T / IPB));   // 4 heads * 64 i-blocks = 256 workgroups
    dim3 block(256);             // 8 wave32 per workgroup
    tsattn_kernel<<<grid, block, 0, stream>>>(q, k1, v1, k2, v2, bu, bv, bw, bg, mask, out);
}
